// RepresentationNetwork_10514079941138
// MI455X (gfx1250) — compile-verified, matrix operand order checked by reading, not run on hardware
//
#include <hip/hip_runtime.h>
#include <hip/hip_bf16.h>
#include <math.h>

// ---------------------------------------------------------------------------
// GAT representation network for gfx1250 (MI455X), wave32 + WMMA.
// GEMMs: v_wmma_f32_16x16x32_f16, f32 accumulate. Weights converted to f16
// and transposed once per launch (contiguous b128 B loads); the A fragment is
// loaded/converted ONCE per wave and reused across both column tiles.
// Edge aggregation is bandwidth/atomic bound; wave-per-edge scatter.
// ---------------------------------------------------------------------------

#define N_NODES 100000
#define N_EDGES 1600000
#define HID 128
#define HEADS 4
#define CPH 32            // channels per head
#define NEG_SLOPE 0.2f
#define LN_EPS 1e-5f

typedef __attribute__((ext_vector_type(16))) _Float16 v16h;
typedef __attribute__((ext_vector_type(8)))  _Float16 v8h;
typedef __attribute__((ext_vector_type(8)))  float    v8f;

// ---------------------------------------------------------------------------
// A-fragment: 16xK f32 row-major -> per-lane v16h per 32-K chunk.
// VGPR layout per CDNA5 ISA 7.12.2 (16-bit A 16x32): lanes 0-15 hold
// M=0..15; elems 0..7 = K-group (0..7 | 8..15 by half-wave), elems 8..15 =
// K-group (16..23 | 24..31).  K compile-time, multiple of 16.
// ---------------------------------------------------------------------------
template <int K> struct AFrag { v16h a[(K + 31) / 32]; };

template <int K>
__device__ inline void load_a(const float* __restrict__ A, int lda,
                              AFrag<K>& fr) {
  const int lane = (int)(threadIdx.x & 31u);
  const int idx  = lane & 15;
  const int half = (lane >> 4) & 1;
  const float* arow = A + (size_t)idx * lda;
#pragma unroll
  for (int c = 0; c < (K + 31) / 32; ++c) {
    const int k0 = c * 32;
    v16h a;
    const float4 f0 = *(const float4*)(arow + k0 + 8 * half);
    const float4 f1 = *(const float4*)(arow + k0 + 8 * half + 4);
    a[0] = (_Float16)f0.x; a[1] = (_Float16)f0.y;
    a[2] = (_Float16)f0.z; a[3] = (_Float16)f0.w;
    a[4] = (_Float16)f1.x; a[5] = (_Float16)f1.y;
    a[6] = (_Float16)f1.z; a[7] = (_Float16)f1.w;
    if (k0 + 16 < K) {   // high K-group present (folds at compile time)
      const float4 f2 = *(const float4*)(arow + k0 + 16 + 8 * half);
      const float4 f3 = *(const float4*)(arow + k0 + 16 + 8 * half + 4);
      a[8]  = (_Float16)f2.x; a[9]  = (_Float16)f2.y;
      a[10] = (_Float16)f2.z; a[11] = (_Float16)f2.w;
      a[12] = (_Float16)f3.x; a[13] = (_Float16)f3.y;
      a[14] = (_Float16)f3.z; a[15] = (_Float16)f3.w;
    } else {
#pragma unroll
      for (int e = 0; e < 8; ++e) a[e + 8] = (_Float16)0.0f;
    }
    fr.a[c] = a;
  }
}

// 16x16 f32 tile = A-frag x Bt-tile.  Bt: f16 pre-transposed [out_col][K] so
// each lane's K-group is one contiguous 16-byte (v8h) load.
template <int K>
__device__ inline v8f wmma_bt(const AFrag<K>& fr,
                              const _Float16* __restrict__ Bt, v8f acc) {
  const int lane = (int)(threadIdx.x & 31u);
  const int idx  = lane & 15;
  const int half = (lane >> 4) & 1;
  const _Float16* brow = Bt + (size_t)idx * K;
#pragma unroll
  for (int c = 0; c < (K + 31) / 32; ++c) {
    const int k0 = c * 32;
    v16h b;
    const v8h blo = *(const v8h*)(brow + k0 + 8 * half);
#pragma unroll
    for (int e = 0; e < 8; ++e) b[e] = blo[e];
    if (k0 + 16 < K) {
      const v8h bhi = *(const v8h*)(brow + k0 + 16 + 8 * half);
#pragma unroll
      for (int e = 0; e < 8; ++e) b[e + 8] = bhi[e];
    } else {
#pragma unroll
      for (int e = 0; e < 8; ++e) b[e + 8] = (_Float16)0.0f;
    }
    acc = __builtin_amdgcn_wmma_f32_16x16x32_f16(
        /*neg_a=*/false, fr.a[c], /*neg_b=*/false, b,
        /*c_mod=*/(short)0, acc, /*reuse_a=*/false, /*reuse_b=*/false);
  }
  return acc;
}

// ---------------------------------------------------------------------------
// Weight prep: Wt[n][k] = (f16) W[k][n].  W: [K,N] f32 -> Wt: [N,K] f16.
// Run once per launch per weight matrix; cost is trivial (<=64KB each).
// ---------------------------------------------------------------------------
__global__ void transpose_to_f16_kernel(const float* __restrict__ W,
                                        _Float16* __restrict__ Wt,
                                        int K, int N) {
  const int i = blockIdx.x * blockDim.x + threadIdx.x;
  if (i >= K * N) return;
  const int k = i / N;
  const int n = i - k * N;
  Wt[(size_t)n * K + k] = (_Float16)W[i];
}

// ---------------------------------------------------------------------------
// Node encoder: h = relu(X@W0+b0)@W1 + b1.  X:[N,48], hidden 128, out 128.
// One block = 16 rows, 128 threads = 4 waves; each wave does 2 of 8 col-tiles.
// ---------------------------------------------------------------------------
__global__ void node_enc_kernel(const float* __restrict__ X,
                                const _Float16* __restrict__ W0t,  // [128][48]
                                const float* __restrict__ b0,
                                const _Float16* __restrict__ W1t,  // [128][128]
                                const float* __restrict__ b1,
                                float* __restrict__ h) {
  __shared__ float hid[16][HID + 4];   // 132 floats/row = 528B, 16B aligned
  const int row0 = blockIdx.x * 16;
  const int wave = (int)(threadIdx.x >> 5);
  const int lane = (int)(threadIdx.x & 31u);
  const int idx  = lane & 15;
  const int half = (lane >> 4) & 1;

  // Stage 1: hidden = relu(X @ W0 + b0), K = 48 (zero-padded to 64)
  {
    AFrag<48> fa;
    load_a<48>(X + (size_t)row0 * 48, 48, fa);
    for (int t = wave; t < 8; t += 4) {
      v8f acc = {};
      acc = wmma_bt<48>(fa, W0t + (size_t)t * 16 * 48, acc);
      const int col = t * 16 + idx;
#pragma unroll
      for (int r = 0; r < 8; ++r)
        hid[r + 8 * half][col] = fmaxf(acc[r] + b0[col], 0.0f);
    }
  }
  __syncthreads();

  // Stage 2: h = hidden @ W1 + b1, K = 128
  {
    AFrag<HID> fa;
    load_a<HID>(&hid[0][0], HID + 4, fa);
    for (int t = wave; t < 8; t += 4) {
      v8f acc = {};
      acc = wmma_bt<HID>(fa, W1t + (size_t)t * 16 * HID, acc);
      const int col = t * 16 + idx;
#pragma unroll
      for (int r = 0; r < 8; ++r)
        h[(size_t)(row0 + r + 8 * half) * HID + col] = acc[r] + b1[col];
    }
  }
}

// ---------------------------------------------------------------------------
// x = h @ W  (128x128, no bias).  One block = 16 rows, 4 waves.
// ---------------------------------------------------------------------------
__global__ void gat_proj_kernel(const float* __restrict__ h,
                                const _Float16* __restrict__ Wt,  // [128][128]
                                float* __restrict__ x) {
  const int row0 = blockIdx.x * 16;
  const int wave = (int)(threadIdx.x >> 5);
  const int lane = (int)(threadIdx.x & 31u);
  const int idx  = lane & 15;
  const int half = (lane >> 4) & 1;
  if (row0 + 16 < N_NODES)   // speculative prefetch of next row tile (L2)
    __builtin_prefetch(h + (size_t)(row0 + 16) * HID + (int)threadIdx.x, 0, 0);
  AFrag<HID> fa;
  load_a<HID>(h + (size_t)row0 * HID, HID, fa);
  for (int t = wave; t < 8; t += 4) {
    v8f acc = {};
    acc = wmma_bt<HID>(fa, Wt + (size_t)t * 16 * HID, acc);
    const int col = t * 16 + idx;
#pragma unroll
    for (int r = 0; r < 8; ++r)
      x[(size_t)(row0 + r + 8 * half) * HID + col] = acc[r];
  }
}

// ---------------------------------------------------------------------------
// Attention logits per (node, head): a_src = <x[n,h,:], att_src[h,:]>, etc.
// ---------------------------------------------------------------------------
__global__ void att_logits_kernel(const float* __restrict__ x,
                                  const float* __restrict__ att_src,
                                  const float* __restrict__ att_dst,
                                  float* __restrict__ asrc,
                                  float* __restrict__ adst) {
  const int i = blockIdx.x * blockDim.x + threadIdx.x;
  if (i >= N_NODES * HEADS) return;
  const int n = i >> 2;
  const int hd = i & 3;
  const float* xv = x + (size_t)n * HID + hd * CPH;
  const float* as = att_src + hd * CPH;
  const float* ad = att_dst + hd * CPH;
  float sa = 0.f, sd = 0.f;
#pragma unroll
  for (int c = 0; c < CPH; ++c) {
    const float v = xv[c];
    sa += v * as[c];
    sd += v * ad[c];
  }
  asrc[i] = sa;
  adst[i] = sd;
}

// ---------------------------------------------------------------------------
// Monotonic float <-> uint encoding so atomicMax(uint) == float max.
// ---------------------------------------------------------------------------
__device__ inline unsigned fenc(float f) {
  unsigned u = __float_as_uint(f);
  return (u >> 31) ? ~u : (u | 0x80000000u);
}
__device__ inline float fdec(unsigned u) {
  return __uint_as_float((u >> 31) ? (u ^ 0x80000000u) : ~u);
}

__global__ void seg_init_kernel(float* __restrict__ agg,
                                unsigned* __restrict__ menc,
                                float* __restrict__ ssum) {
  const int i = blockIdx.x * blockDim.x + threadIdx.x;
  if (i < N_NODES * HID) agg[i] = 0.0f;
  if (i < N_NODES * HEADS) { menc[i] = 0u; ssum[i] = 0.0f; }
}

// Edge pass 1: segment max of leaky-relu logits (self-loops are virtual
// edges e in [E, E+N)).  One thread per edge, 4 heads each.
__global__ void edge_max_kernel(const int* __restrict__ ei,
                                const float* __restrict__ asrc,
                                const float* __restrict__ adst,
                                unsigned* __restrict__ menc) {
  const int e = blockIdx.x * blockDim.x + threadIdx.x;
  if (e >= N_EDGES + N_NODES) return;
  int s, d;
  if (e < N_EDGES) { s = ei[e]; d = ei[N_EDGES + e]; }
  else             { s = d = e - N_EDGES; }
#pragma unroll
  for (int hd = 0; hd < HEADS; ++hd) {
    float v = asrc[s * HEADS + hd] + adst[d * HEADS + hd];
    v = (v > 0.0f) ? v : NEG_SLOPE * v;
    atomicMax(&menc[d * HEADS + hd], fenc(v));
  }
}

// Edge pass 2: w = exp(e - m[dst]); ssum[dst] += w; agg[dst] += w * x[src].
// One wave per edge: lane owns 4 channels (float4), head = lane>>3.
__global__ void edge_accum_kernel(const int* __restrict__ ei,
                                  const float* __restrict__ asrc,
                                  const float* __restrict__ adst,
                                  const unsigned* __restrict__ menc,
                                  const float* __restrict__ x,
                                  float* __restrict__ ssum,
                                  float* __restrict__ agg) {
  const int e = blockIdx.x * 8 + (int)(threadIdx.x >> 5);
  if (e >= N_EDGES + N_NODES) return;
  const int lane = (int)(threadIdx.x & 31u);
  int s, d;
  if (e < N_EDGES) { s = ei[e]; d = ei[N_EDGES + e]; }
  else             { s = d = e - N_EDGES; }
  const int hd = lane >> 3;
  float lg = asrc[s * HEADS + hd] + adst[d * HEADS + hd];
  lg = (lg > 0.0f) ? lg : NEG_SLOPE * lg;
  const float m = fdec(menc[d * HEADS + hd]);
  const float w = __expf(lg - m);
  if ((lane & 7) == 0) atomicAdd(&ssum[d * HEADS + hd], w);
  const float4 xv = *(const float4*)(x + (size_t)s * HID + lane * 4);
  float* ap = agg + (size_t)d * HID + lane * 4;
  atomicAdd(ap + 0, w * xv.x);
  atomicAdd(ap + 1, w * xv.y);
  atomicAdd(ap + 2, w * xv.z);
  atomicAdd(ap + 3, w * xv.w);
}

// ---------------------------------------------------------------------------
// Per node: out = agg/ssum + bias + h_res; h = relu(LayerNorm(out)).
// Block = node, 128 threads, LDS tree reductions for mean/var.
// ---------------------------------------------------------------------------
__global__ void finalize_kernel(const float* __restrict__ agg,
                                const float* __restrict__ ssum,
                                const float* __restrict__ bias,
                                const float* __restrict__ gamma,
                                const float* __restrict__ beta,
                                float* __restrict__ h) {
  __shared__ float red[HID];
  const int n = blockIdx.x;
  const int c = (int)threadIdx.x;
  const float denom = ssum[n * HEADS + (c >> 5)];
  const float val = agg[(size_t)n * HID + c] / denom + bias[c] +
                    h[(size_t)n * HID + c];
  red[c] = val;
  __syncthreads();
  for (int s = 64; s > 0; s >>= 1) {
    if (c < s) red[c] += red[c + s];
    __syncthreads();
  }
  const float mu = red[0] * (1.0f / HID);
  __syncthreads();
  const float dv = val - mu;
  red[c] = dv * dv;
  __syncthreads();
  for (int s = 64; s > 0; s >>= 1) {
    if (c < s) red[c] += red[c + s];
    __syncthreads();
  }
  const float var = red[0] * (1.0f / HID);
  const float o = dv * rsqrtf(var + LN_EPS) * gamma[c] + beta[c];
  h[(size_t)n * HID + c] = fmaxf(o, 0.0f);
}

// ---------------------------------------------------------------------------
// Pipe MLP: q = relu(pf @ W0 + b0) @ W1 + b1, pf = [pipeline_state(8), rp(1)]
// ---------------------------------------------------------------------------
__global__ void pipe_mlp_kernel(const float* __restrict__ ps,
                                const float* __restrict__ rp,
                                const float* __restrict__ W0,
                                const float* __restrict__ b0,
                                const float* __restrict__ W1,
                                const float* __restrict__ b1,
                                float* __restrict__ q) {
  __shared__ float pf[9];
  __shared__ float hid[HID];
  const int t = (int)threadIdx.x;
  if (t < 8) pf[t] = ps[t];
  if (t == 8) pf[8] = rp[0];
  __syncthreads();
  float acc = b0[t];
#pragma unroll
  for (int i = 0; i < 9; ++i) acc += pf[i] * W0[i * HID + t];
  hid[t] = fmaxf(acc, 0.0f);
  __syncthreads();
  float acc2 = b1[t];
  for (int k = 0; k < HID; ++k) acc2 += hid[k] * W1[k * HID + t];
  q[t] = acc2;
}

// ---------------------------------------------------------------------------
// Host launcher.  Input order (setup_inputs insertion order, params flattened
// depth-first in insertion order):
//  0 node_features [100000,48]      1 edge_index [2,1600000] int32
//  2 edge_attr (unused)             3 pipeline_state [1,8]
//  4 register_pressure [1,1]        5 ready_mask / 6 scheduled_mask (unused)
//  7..10  node_enc: W0,b0,W1,b1     11..14 pipe_enc: W0,b0,W1,b1
//  15+4i  gat[i]: W, att_src, att_dst, bias   (i=0..2)
//  27+2i  ln[i]: gamma, beta                  (i=0..2)
// d_out: h [100000*128] then q [128].
// ---------------------------------------------------------------------------
extern "C" void kernel_launch(void* const* d_in, const int* in_sizes, int n_in,
                              void* d_out, int out_size, void* d_ws, size_t ws_size,
                              hipStream_t stream) {
  const int N = N_NODES, E = N_EDGES;

  const float* node_features = (const float*)d_in[0];
  const int*   edge_index    = (const int*)d_in[1];
  const float* pipeline_st   = (const float*)d_in[3];
  const float* reg_pressure  = (const float*)d_in[4];

  const float* ne0W = (const float*)d_in[7];
  const float* ne0b = (const float*)d_in[8];
  const float* ne1W = (const float*)d_in[9];
  const float* ne1b = (const float*)d_in[10];
  const float* pe0W = (const float*)d_in[11];
  const float* pe0b = (const float*)d_in[12];
  const float* pe1W = (const float*)d_in[13];
  const float* pe1b = (const float*)d_in[14];

  const float *gW[3], *gAs[3], *gAd[3], *gB[3], *lnG[3], *lnB[3];
  for (int i = 0; i < 3; ++i) {
    gW[i]  = (const float*)d_in[15 + 4 * i];
    gAs[i] = (const float*)d_in[16 + 4 * i];
    gAd[i] = (const float*)d_in[17 + 4 * i];
    gB[i]  = (const float*)d_in[18 + 4 * i];
    lnG[i] = (const float*)d_in[27 + 2 * i];
    lnB[i] = (const float*)d_in[28 + 2 * i];
  }

  float* h = (float*)d_out;              // [N,128], final output region
  float* q = h + (size_t)N * HID;        // [128]

  // Workspace: x[N*128] agg[N*128] asrc[N*4] adst[N*4] menc[N*4] ssum[N*4]
  // then f16 transposed weights (16B-aligned: 27.2M floats = 108.8MB offset).
  float*    x    = (float*)d_ws;
  float*    agg  = x + (size_t)N * HID;
  float*    asrc = agg + (size_t)N * HID;
  float*    adst = asrc + (size_t)N * HEADS;
  unsigned* menc = (unsigned*)(adst + (size_t)N * HEADS);
  float*    ssum = (float*)(menc + (size_t)N * HEADS);
  _Float16* ne0Wt = (_Float16*)(ssum + (size_t)N * HEADS);   // [128][48]
  _Float16* ne1Wt = ne0Wt + 128 * 48;                        // [128][128]
  _Float16* gWt[3];
  gWt[0] = ne1Wt + 128 * 128;
  gWt[1] = gWt[0] + 128 * 128;
  gWt[2] = gWt[1] + 128 * 128;

  const int rowBlocks = N / 16;                 // 100000 % 16 == 0
  const int edgeTotal = E + N;                  // real edges + self loops

  // One-time (per launch) weight conversion: f32 [K,N] -> f16 [N,K].
  transpose_to_f16_kernel<<<(48 * 128 + 255) / 256, 256, 0, stream>>>(ne0W, ne0Wt, 48, 128);
  transpose_to_f16_kernel<<<(128 * 128 + 255) / 256, 256, 0, stream>>>(ne1W, ne1Wt, 128, 128);
  for (int l = 0; l < 3; ++l)
    transpose_to_f16_kernel<<<(128 * 128 + 255) / 256, 256, 0, stream>>>(gW[l], gWt[l], 128, 128);

  node_enc_kernel<<<rowBlocks, 128, 0, stream>>>(node_features, ne0Wt, ne0b,
                                                 ne1Wt, ne1b, h);

  for (int l = 0; l < 3; ++l) {
    gat_proj_kernel<<<rowBlocks, 128, 0, stream>>>(h, gWt[l], x);
    att_logits_kernel<<<(N * HEADS + 255) / 256, 256, 0, stream>>>(
        x, gAs[l], gAd[l], asrc, adst);
    seg_init_kernel<<<(N * HID + 255) / 256, 256, 0, stream>>>(agg, menc, ssum);
    edge_max_kernel<<<(edgeTotal + 255) / 256, 256, 0, stream>>>(
        edge_index, asrc, adst, menc);
    edge_accum_kernel<<<(edgeTotal + 7) / 8, 256, 0, stream>>>(
        edge_index, asrc, adst, menc, x, ssum, agg);
    finalize_kernel<<<N, HID, 0, stream>>>(agg, ssum, gB[l], lnG[l], lnB[l], h);
  }

  pipe_mlp_kernel<<<1, HID, 0, stream>>>(pipeline_st, reg_pressure,
                                         pe0W, pe0b, pe1W, pe1b, q);
}
